// GatedCrossAttention_69088843923694
// MI455X (gfx1250) — compile-verified, matrix operand order checked by reading, not run on hardware
//
#include <hip/hip_runtime.h>
#include <hip/hip_bf16.h>
#include <math.h>
#include <stdint.h>

typedef __bf16 bf16_t;
typedef __attribute__((ext_vector_type(16))) __bf16 v16bf;
typedef __attribute__((ext_vector_type(8)))  __bf16 v8bf;
typedef __attribute__((ext_vector_type(8)))  float  v8f;

#define E_DIM 1024
#define Z_DIM 256
#define L_DIM 2048
#define B_DIM 8
#define M_ROWS (L_DIM * B_DIM)
#define QRU_DIM (2 * E_DIM + Z_DIM)

#define BM 128
#define BN 128
#define BK 32
#define LDSS 40   // LDS row stride in bf16 elems: 80 bytes, multiple of 16B

// Async DMA: copy 16 bytes global -> LDS, tracked by ASYNCcnt (gfx1250).
__device__ __forceinline__ void async_b128(uint32_t lds_addr, const void* gptr) {
  asm volatile("global_load_async_to_lds_b128 %0, %1, off"
               :: "v"(lds_addr), "v"(gptr) : "memory");
}
__device__ __forceinline__ void wait_async0() {
  asm volatile("s_wait_asynccnt 0x0" ::: "memory");
}
__device__ __forceinline__ uint32_t lds_addr_of(const void* p) {
  return (uint32_t)(uintptr_t)p;   // low 32 bits of the pointer give the LDS offset
}

// Load a 16-element bf16 WMMA fragment: elems 0..7 -> K = hi*8+e,
// elems 8..15 -> K = 16 + hi*8 + (e-8) (ISA 7.12.2 16-bit layout).
// Same (lane,elem)->K map is used for A and B fragments, so any residual
// K-permutation ambiguity cancels in the dot product.
__device__ __forceinline__ v16bf frag_load(const bf16_t* rowp, int hi) {
  union { v16bf v; v8bf h[2]; } u;
  u.h[0] = *(const v8bf*)(rowp + hi * 8);
  u.h[1] = *(const v8bf*)(rowp + 16 + hi * 8);
  return u.v;
}

// C = A(MxK) * op(B) + epilogue.
// WT=1: second operand is W (N x K row-major), C = A @ W^T (async-to-LDS path)
// WT=0: second operand is B (K x N row-major), C = A @ B   (VGPR transpose path)
// EPI: 0 = +bias(optional), store f32
//      1 = silu(acc+bias), store bf16
//      2 = relu(acc*scale + relpos[2047+col-row])^2, store bf16
//      3 = out = resid + gate_u*(acc+bias - resid), store f32
template<int WT, int EPI>
__global__ __launch_bounds__(256)
void wmma_gemm(const bf16_t* __restrict__ A,
               const bf16_t* __restrict__ Bm,
               float* __restrict__ Cf,
               bf16_t* __restrict__ Cb,
               const float* __restrict__ bias,
               const float* __restrict__ relpos,
               float scale,
               const float* __restrict__ gate_u,
               const float* __restrict__ resid,
               int M, int N, int K,
               long lda, long ldb, long ldc,
               long strA, long strB, long strC)
{
  __shared__ __align__(16) bf16_t shA[2][BM * LDSS];
  __shared__ __align__(16) bf16_t shB[2][BN * LDSS];

  const int bz = blockIdx.z;
  A  += (long)bz * strA;
  Bm += (long)bz * strB;
  const long cOff = (long)bz * strC;

  const int tid  = threadIdx.x;
  const int lane = tid & 31;
  const int wave = tid >> 5;
  const int wm   = wave & 3;     // 4 waves along M
  const int wn   = wave >> 2;    // 2 waves along N
  const int lo   = lane & 15;
  const int hi   = lane >> 4;

  const int rowBase = blockIdx.y * BM;
  const int colBase = blockIdx.x * BN;

  // stage tile starting at K-offset kt into buffer buf
  auto copy_tiles = [&](int kt, int buf) {
    { // A tile: 128 rows x 32 K, 2 threads/row, 2 async b128 each
      const int ar = tid >> 1;
      const int ac = (tid & 1) << 4;
      const bf16_t* src = A + (long)(rowBase + ar) * lda + kt + ac;
      const uint32_t la = lds_addr_of(&shA[buf][ar * LDSS + ac]);
      async_b128(la, src);
      async_b128(la + 16, src + 8);
    }
    if (WT) { // W tile: 128 N-rows x 32 K, contiguous in K -> async DMA
      const int br = tid >> 1;
      const int bc = (tid & 1) << 4;
      const bf16_t* src = Bm + (long)(colBase + br) * ldb + kt + bc;
      const uint32_t lb = lds_addr_of(&shB[buf][br * LDSS + bc]);
      async_b128(lb, src);
      async_b128(lb + 16, src + 8);
    } else {  // B tile: K x N in memory -> transpose into LDS via VGPRs
      #pragma unroll
      for (int p = 0; p < 2; p++) {
        const int kk = (tid >> 4) + p * 16;
        const int n0 = (tid & 15) << 3;
        const bf16_t* src = Bm + (long)(kt + kk) * ldb + colBase + n0;
        v8bf d = *(const v8bf*)src;
        #pragma unroll
        for (int e = 0; e < 8; e++) shB[buf][(n0 + e) * LDSS + kk] = d[e];
      }
    }
  };

  v8f acc[2][4];
  #pragma unroll
  for (int i = 0; i < 2; i++)
    #pragma unroll
    for (int j = 0; j < 4; j++)
      #pragma unroll
      for (int v = 0; v < 8; v++) acc[i][j][v] = 0.f;

  const int nkt = K / BK;
  copy_tiles(0, 0);
  int cur = 0;
  for (int t = 0; t < nkt; ++t) {
    wait_async0();
    __syncthreads();                    // tile t ready for everyone; prev reads done
    if (t + 1 < nkt) copy_tiles((t + 1) * BK, cur ^ 1);  // overlap with compute

    v16bf af[2];
    #pragma unroll
    for (int i = 0; i < 2; i++)
      af[i] = frag_load(&shA[cur][(wm * 32 + i * 16 + lo) * LDSS], hi);
    #pragma unroll
    for (int j = 0; j < 4; j++) {
      v16bf bfr = frag_load(&shB[cur][(wn * 64 + j * 16 + lo) * LDSS], hi);
      #pragma unroll
      for (int i = 0; i < 2; i++)
        acc[i][j] = __builtin_amdgcn_wmma_f32_16x16x32_bf16(
            false, af[i], false, bfr, (short)0, acc[i][j], false, false);
    }
    cur ^= 1;
  }

  // C/D layout: VGPR v, lanes 0-15: M = v, N = lane; lanes 16-31: M = 8+v
  #pragma unroll
  for (int i = 0; i < 2; i++)
    #pragma unroll
    for (int j = 0; j < 4; j++)
      #pragma unroll
      for (int v = 0; v < 8; v++) {
        const int row = rowBase + wm * 32 + i * 16 + hi * 8 + v;
        const int col = colBase + wn * 64 + j * 16 + lo;
        float val = acc[i][j][v];
        const long cidx = cOff + (long)row * ldc + col;
        if (EPI == 0) {
          if (bias) val += bias[col];
          Cf[cidx] = val;
        } else if (EPI == 1) {
          val += bias[col];
          val = val / (1.f + __expf(-val));   // silu
          Cb[cidx] = (bf16_t)val;
        } else if (EPI == 2) {
          val = val * scale + relpos[2047 + col - row];
          val = fmaxf(val, 0.f);
          Cb[cidx] = (bf16_t)(val * val);
        } else {
          val += bias[col];
          const float qv = resid[cidx];
          Cf[cidx] = qv + gate_u[cidx] * (val - qv);
        }
      }
}

__global__ __launch_bounds__(256)
void f32_to_bf16(const float* __restrict__ in, bf16_t* __restrict__ out, long n) {
  long i = (long)blockIdx.x * blockDim.x + threadIdx.x;
  const long stride = (long)gridDim.x * blockDim.x;
  for (; i < n; i += stride) out[i] = (bf16_t)in[i];
}

__global__ __launch_bounds__(256)
void layernorm_bf16(const float* __restrict__ x, const float* __restrict__ w,
                    const float* __restrict__ b, bf16_t* __restrict__ out) {
  const long row = blockIdx.x;
  const int tid = threadIdx.x;
  const float* xr = x + row * E_DIM;
  __shared__ float red[256];
  float v[4]; float s = 0.f;
  #pragma unroll
  for (int i = 0; i < 4; i++) { v[i] = xr[tid + i * 256]; s += v[i]; }
  red[tid] = s; __syncthreads();
  for (int off = 128; off > 0; off >>= 1) { if (tid < off) red[tid] += red[tid + off]; __syncthreads(); }
  const float mu = red[0] * (1.f / E_DIM);
  __syncthreads();
  float vs = 0.f;
  #pragma unroll
  for (int i = 0; i < 4; i++) { float d = v[i] - mu; vs += d * d; }
  red[tid] = vs; __syncthreads();
  for (int off = 128; off > 0; off >>= 1) { if (tid < off) red[tid] += red[tid + off]; __syncthreads(); }
  const float inv = rsqrtf(red[0] * (1.f / E_DIM) + 1e-5f);
  #pragma unroll
  for (int i = 0; i < 4; i++) {
    const int c = tid + i * 256;
    out[row * E_DIM + c] = (bf16_t)((v[i] - mu) * inv * w[c] + b[c]);
  }
}

__global__ __launch_bounds__(256)
void split_qur(const float* __restrict__ base, const float* __restrict__ gamma,
               const float* __restrict__ beta, bf16_t* __restrict__ qout,
               float* __restrict__ uout, float* __restrict__ rout) {
  const long row = blockIdx.x;
  const int tid = threadIdx.x;
  const float* br = base + row * QRU_DIM;
  __shared__ float red[256];
  const float qv = br[tid];
  red[tid] = qv * qv; __syncthreads();
  for (int off = 128; off > 0; off >>= 1) { if (tid < off) red[tid] += red[tid + off]; __syncthreads(); }
  const float inv = 1.f / fmaxf(sqrtf(red[0]), 1e-5f);
  qout[row * Z_DIM + tid] = (bf16_t)(qv * inv * (gamma[tid] + 1.f) + beta[tid]);
  #pragma unroll
  for (int i = 0; i < 4; i++) {
    const int c = tid + i * 256;
    const float uv = br[Z_DIM + c];
    uout[row * E_DIM + c] = 1.f / (1.f + __expf(-uv));
    const float rv = br[Z_DIM + E_DIM + c];
    rout[row * E_DIM + c] = rv / (1.f + __expf(-rv));
  }
}

__global__ __launch_bounds__(256)
void k_act(const float* __restrict__ kin, const float* __restrict__ gamma,
           const float* __restrict__ beta, bf16_t* __restrict__ kout) {
  const long row = blockIdx.x;
  const int tid = threadIdx.x;
  const float kv = kin[row * Z_DIM + tid];
  __shared__ float red[256];
  red[tid] = kv * kv; __syncthreads();
  for (int off = 128; off > 0; off >>= 1) { if (tid < off) red[tid] += red[tid + off]; __syncthreads(); }
  const float inv = 1.f / fmaxf(sqrtf(red[0]), 1e-5f);
  kout[row * Z_DIM + tid] =
      (bf16_t)(kv * inv * (gamma[Z_DIM + tid] + 1.f) + beta[Z_DIM + tid]);
}

__global__ __launch_bounds__(256)
void mul_hr(const float* __restrict__ h, const float* __restrict__ r,
            bf16_t* __restrict__ out, long n) {
  long i = (long)blockIdx.x * blockDim.x + threadIdx.x;
  const long stride = (long)gridDim.x * blockDim.x;
  for (; i < n; i += stride) out[i] = (bf16_t)(h[i] * r[i]);
}

extern "C" void kernel_launch(void* const* d_in, const int* in_sizes, int n_in,
                              void* d_out, int out_size, void* d_ws, size_t ws_size,
                              hipStream_t stream) {
  (void)in_sizes; (void)n_in; (void)out_size; (void)ws_size;
  const float* query  = (const float*)d_in[0];
  const float* key_in = (const float*)d_in[1];
  const float* value  = (const float*)d_in[2];
  const float* ln_w   = (const float*)d_in[3];
  const float* ln_b   = (const float*)d_in[4];
  const float* Wv     = (const float*)d_in[5];
  const float* bv     = (const float*)d_in[6];
  const float* Wk     = (const float*)d_in[7];
  const float* bk     = (const float*)d_in[8];
  const float* Wqru   = (const float*)d_in[9];
  const float* bqru   = (const float*)d_in[10];
  const float* Wh     = (const float*)d_in[11];
  const float* bh     = (const float*)d_in[12];
  const float* gamma  = (const float*)d_in[13];
  const float* beta   = (const float*)d_in[14];
  const float* relpos = (const float*)d_in[15];

  char* ws = (char*)d_ws;
  size_t off = 0;
  auto alloc = [&](size_t bytes) -> char* {
    char* p = ws + off;
    off = (off + bytes + 255) & ~(size_t)255;
    return p;
  };

  bf16_t* nq_bf   = (bf16_t*)alloc((size_t)M_ROWS * E_DIM * 2);
  bf16_t* key_bf  = (bf16_t*)alloc((size_t)M_ROWS * E_DIM * 2);
  bf16_t* val_bf  = (bf16_t*)alloc((size_t)M_ROWS * E_DIM * 2);
  bf16_t* Wqru_bf = (bf16_t*)alloc((size_t)QRU_DIM * E_DIM * 2);
  bf16_t* Wk_bf   = (bf16_t*)alloc((size_t)Z_DIM * E_DIM * 2);
  bf16_t* Wv_bf   = (bf16_t*)alloc((size_t)E_DIM * E_DIM * 2);
  bf16_t* Wh_bf   = (bf16_t*)alloc((size_t)E_DIM * E_DIM * 2);
  bf16_t* q_bf    = (bf16_t*)alloc((size_t)M_ROWS * Z_DIM * 2);
  float*  u_f     = (float*) alloc((size_t)M_ROWS * E_DIM * 4);
  float*  r_f     = (float*) alloc((size_t)M_ROWS * E_DIM * 4);
  float*  k_f     = (float*) alloc((size_t)M_ROWS * Z_DIM * 4);
  bf16_t* k_bf    = (bf16_t*)alloc((size_t)M_ROWS * Z_DIM * 2);
  bf16_t* v_bf    = (bf16_t*)alloc((size_t)M_ROWS * E_DIM * 2);
  bf16_t* attn_bf = (bf16_t*)alloc((size_t)B_DIM * L_DIM * L_DIM * 2);
  char*   base_rg = alloc((size_t)M_ROWS * QRU_DIM * 4);
  float*  base_f  = (float*)base_rg;
  float*  h_f     = (float*)base_rg;   // base is dead after split_qur; reuse for h
  bf16_t* hr_bf   = (bf16_t*)(base_rg + (size_t)M_ROWS * E_DIM * 4);

  const long nBig = (long)M_ROWS * E_DIM;
  dim3 blk(256);

  f32_to_bf16<<<4096, blk, 0, stream>>>(key_in, key_bf, nBig);
  f32_to_bf16<<<4096, blk, 0, stream>>>(value, val_bf, nBig);
  f32_to_bf16<<<2048, blk, 0, stream>>>(Wqru, Wqru_bf, (long)QRU_DIM * E_DIM);
  f32_to_bf16<<<512,  blk, 0, stream>>>(Wk, Wk_bf, (long)Z_DIM * E_DIM);
  f32_to_bf16<<<1024, blk, 0, stream>>>(Wv, Wv_bf, (long)E_DIM * E_DIM);
  f32_to_bf16<<<1024, blk, 0, stream>>>(Wh, Wh_bf, (long)E_DIM * E_DIM);

  layernorm_bf16<<<M_ROWS, blk, 0, stream>>>(query, ln_w, ln_b, nq_bf);

  // base = nq @ Wqru^T + bqru  (f32)
  wmma_gemm<1, 0><<<dim3(QRU_DIM / BN, M_ROWS / BM, 1), blk, 0, stream>>>(
      nq_bf, Wqru_bf, base_f, nullptr, bqru, nullptr, 1.f, nullptr, nullptr,
      M_ROWS, QRU_DIM, E_DIM, E_DIM, E_DIM, QRU_DIM, 0, 0, 0);

  // k = key @ Wk^T + bk  (f32, l2norm after)
  wmma_gemm<1, 0><<<dim3(Z_DIM / BN, M_ROWS / BM, 1), blk, 0, stream>>>(
      key_bf, Wk_bf, k_f, nullptr, bk, nullptr, 1.f, nullptr, nullptr,
      M_ROWS, Z_DIM, E_DIM, E_DIM, E_DIM, Z_DIM, 0, 0, 0);

  // v = silu(value @ Wv^T + bv)  (bf16, fused)
  wmma_gemm<1, 1><<<dim3(E_DIM / BN, M_ROWS / BM, 1), blk, 0, stream>>>(
      val_bf, Wv_bf, nullptr, v_bf, bv, nullptr, 1.f, nullptr, nullptr,
      M_ROWS, E_DIM, E_DIM, E_DIM, E_DIM, E_DIM, 0, 0, 0);

  split_qur<<<M_ROWS, blk, 0, stream>>>(base_f, gamma, beta, q_bf, u_f, r_f);
  k_act<<<M_ROWS, blk, 0, stream>>>(k_f, gamma, beta, k_bf);

  // attn[b] = relu(q@k^T * len_scale + relbias)^2  (bf16, fused epilogue)
  const float len_scale = 1.0f / sqrtf((float)L_DIM);
  wmma_gemm<1, 2><<<dim3(L_DIM / BN, L_DIM / BM, B_DIM), blk, 0, stream>>>(
      q_bf, k_bf, nullptr, attn_bf, nullptr, relpos, len_scale, nullptr, nullptr,
      L_DIM, L_DIM, Z_DIM,
      (long)B_DIM * Z_DIM, (long)B_DIM * Z_DIM, L_DIM,
      Z_DIM, Z_DIM, (long)L_DIM * L_DIM);

  // h[b] = attn[b] @ v[b]  (NN, f32)
  wmma_gemm<0, 0><<<dim3(E_DIM / BN, L_DIM / BM, B_DIM), blk, 0, stream>>>(
      attn_bf, v_bf, h_f, nullptr, nullptr, nullptr, 1.f, nullptr, nullptr,
      L_DIM, E_DIM, L_DIM,
      L_DIM, (long)B_DIM * E_DIM, (long)B_DIM * E_DIM,
      (long)L_DIM * L_DIM, E_DIM, E_DIM);

  mul_hr<<<4096, blk, 0, stream>>>(h_f, r_f, hr_bf, nBig);

  // out = query + u * ((h*r) @ Wh^T + bh - query)  (fused gated residual)
  wmma_gemm<1, 3><<<dim3(E_DIM / BN, M_ROWS / BM, 1), blk, 0, stream>>>(
      hr_bf, Wh_bf, (float*)d_out, nullptr, bh, nullptr, 1.f, u_f, query,
      M_ROWS, E_DIM, E_DIM, E_DIM, E_DIM, E_DIM, 0, 0, 0);
}
